// MixtralDecoderLayer_39711267618828
// MI455X (gfx1250) — matrix operand order repaired and checked by reading, not compiled, not run on hardware
//
#include <hip/hip_runtime.h>
#include <stdint.h>

// ---------------- problem constants ----------------
#define TKN   4096      // B*S tokens
#define BATCH 2
#define SEQ   2048
#define HDIM  1024      // hidden
#define QDIM  1024      // NH*HD
#define KVDIM 256       // NKV*HD
#define NHEAD 16
#define NKVH  4
#define HSZ   64
#define NEXP  8
#define IDIM  2048
#define NASSIGN 8192    // TKN * top2

typedef __bf16 v16bf __attribute__((ext_vector_type(16)));
typedef __bf16 v8bf  __attribute__((ext_vector_type(8)));
typedef float  v8f   __attribute__((ext_vector_type(8)));

union AFrag { v16bf v; v8bf h[2]; unsigned short s[16]; };
union BFrag { v16bf v; unsigned short s[16]; };

__device__ inline unsigned short f2bf(float f) {
  union { float f; unsigned u; } v; v.f = f;
  unsigned r = v.u + 0x7FFFu + ((v.u >> 16) & 1u);
  return (unsigned short)(r >> 16);
}
__device__ inline float bf2f(unsigned short h) {
  union { unsigned u; float f; } v; v.u = ((unsigned)h) << 16;
  return v.f;
}
__device__ inline v8f zero8() {
  v8f z;
#pragma unroll
  for (int i = 0; i < 8; ++i) z[i] = 0.f;
  return z;
}

// ---------------- CDNA5 async global->LDS staging ----------------
#if __has_builtin(__builtin_amdgcn_global_load_async_to_lds_b128) && \
    __has_builtin(__builtin_amdgcn_s_wait_asynccnt)
#define MX_ASYNC 1
#endif

typedef int mx_v4i __attribute__((vector_size(16)));
typedef __attribute__((address_space(1))) mx_v4i mx_gv4i;  // global
typedef __attribute__((address_space(3))) mx_v4i mx_lv4i;  // LDS

// copy 16 bytes (8 bf16) global -> LDS (async DMA path on CDNA5)
__device__ inline void mx_cp16(const unsigned short* g, unsigned short* l) {
#ifdef MX_ASYNC
  __builtin_amdgcn_global_load_async_to_lds_b128((mx_gv4i*)g, (mx_lv4i*)l, 0, 0);
#else
  *(v8bf*)l = *(const v8bf*)g;
#endif
}
template <int N> __device__ inline void mx_wait_stage() {
#ifdef MX_ASYNC
  __builtin_amdgcn_s_wait_asynccnt(N);
#endif
}

// ---------------- fp32 -> bf16 transposed weight conversion ----------------
// in: [e][R][C] fp32   out: [e][C][R] bf16
__global__ void mx_convT(const float* __restrict__ in, unsigned short* __restrict__ out,
                         int R, int C) {
  int e = blockIdx.z;
  size_t base = (size_t)e * R * C;
  int idx = blockIdx.x * 256 + threadIdx.x;
  if (idx >= R * C) return;
  int r = idx / C, c = idx % C;
  out[base + (size_t)c * R + r] = f2bf(in[base + idx]);
}

// ---------------- RMSNorm (fp32 in, bf16 out) ----------------
__global__ void __launch_bounds__(256) mx_rmsnorm(const float* __restrict__ x,
                                                  const float* __restrict__ w,
                                                  unsigned short* __restrict__ out) {
  int t = blockIdx.x;
  __shared__ float red[256];
  float ss = 0.f;
  for (int i = threadIdx.x; i < HDIM; i += 256) {
    float v = x[(size_t)t * HDIM + i];
    ss += v * v;
  }
  red[threadIdx.x] = ss;
  __syncthreads();
  for (int s = 128; s > 0; s >>= 1) {
    if (threadIdx.x < s) red[threadIdx.x] += red[threadIdx.x + s];
    __syncthreads();
  }
  float inv = rsqrtf(red[0] / (float)HDIM + 1e-6f);
  for (int i = threadIdx.x; i < HDIM; i += 256)
    out[(size_t)t * HDIM + i] = f2bf(x[(size_t)t * HDIM + i] * inv * w[i]);
}

// ---------------- dense GEMM: C[M,N](f32) = A[M,K](bf16) x Bt[N,K](bf16)^T ----------------
// 256 threads (8 waves), tile 128M x 64N, K step 64, double-buffered async LDS,
// peeled last iteration (straight-line steady state).
__global__ void __launch_bounds__(256) mx_gemm(const unsigned short* __restrict__ A,
                                               const unsigned short* __restrict__ Bt,
                                               float* __restrict__ C,
                                               const float* __restrict__ resid,
                                               int M, int N, int Kd) {
  __shared__ __align__(64) unsigned short As[2][128 * 64];
  __shared__ __align__(64) unsigned short Bs[2][64 * 64];
  const int m0 = blockIdx.x * 128;
  const int n0 = blockIdx.y * 64;
  const int tid = threadIdx.x;
  const int wvid = tid >> 5, lane = tid & 31;
  const int ln = lane & 15, hf = lane >> 4;
  v8f acc[4];
#pragma unroll
  for (int i = 0; i < 4; ++i) acc[i] = zero8();

  const int ar = tid >> 1, ac = (tid & 1) * 32;
  const int br = tid >> 2, bc = (tid & 3) * 16;
  const unsigned short* gA = A + (size_t)(m0 + ar) * Kd + ac;
  const unsigned short* gB = Bt + (size_t)(n0 + br) * Kd + bc;
  auto stage = [&](int buf, int k0) {
#pragma unroll
    for (int j = 0; j < 4; ++j)
      mx_cp16(gA + k0 + j * 8, &As[buf][ar * 64 + ac + j * 8]);
    mx_cp16(gB + k0, &Bs[buf][br * 64 + bc]);
    mx_cp16(gB + k0 + 8, &Bs[buf][br * 64 + bc + 8]);
  };
  auto compute = [&](int buf) {
    int arow = wvid * 16 + ln;
#pragma unroll
    for (int kk = 0; kk < 2; ++kk) {
      AFrag a;
      a.h[0] = *(const v8bf*)&As[buf][arow * 64 + kk * 32 + hf * 16];
      a.h[1] = *(const v8bf*)&As[buf][arow * 64 + kk * 32 + hf * 16 + 8];
#pragma unroll
      for (int nt = 0; nt < 4; ++nt) {
        BFrag b;
        b.v = *(const v16bf*)&Bs[buf][(nt * 16 + ln) * 64 + kk * 32 + hf * 16];
        acc[nt] = __builtin_amdgcn_wmma_f32_16x16x32_bf16(false, a.v, false, b.v,
                                                          (short)0, acc[nt], false, false);
      }
    }
  };
  stage(0, 0);
  const int nk = Kd / 64;
  for (int ki = 0; ki < nk - 1; ++ki) {
    stage((ki + 1) & 1, (ki + 1) * 64);
    mx_wait_stage<6>();
    __syncthreads();
    compute(ki & 1);
    __syncthreads();
  }
  mx_wait_stage<0>();
  __syncthreads();
  compute((nk - 1) & 1);

  int row0 = m0 + wvid * 16 + hf * 8;
#pragma unroll
  for (int nt = 0; nt < 4; ++nt) {
#pragma unroll
    for (int r = 0; r < 8; ++r) {
      size_t idx = (size_t)(row0 + r) * N + n0 + nt * 16 + ln;
      float v = acc[nt][r];
      if (resid) v += resid[idx];
      C[idx] = v;
    }
  }
}

// ---------------- RoPE + fp32->bf16 for q/k/v ----------------
__global__ void __launch_bounds__(256) mx_rope(const float* __restrict__ qp,
                                               const float* __restrict__ kp,
                                               const float* __restrict__ vp,
                                               const int* __restrict__ pos,
                                               unsigned short* __restrict__ qo,
                                               unsigned short* __restrict__ ko,
                                               unsigned short* __restrict__ vo) {
  int t = blockIdx.x;
  float p = (float)pos[t];
  const float LN_THETA = 9.210340371976184f; // ln(10000)
  for (int i = threadIdx.x; i < QDIM; i += 256) {
    int d = i & 63, ii = d & 31;
    float ang = p * __expf(-(float)(2 * ii) * (LN_THETA / 64.f));
    float c = __cosf(ang), s = __sinf(ang);
    float x = qp[(size_t)t * QDIM + i];
    float xr = (d < 32) ? -qp[(size_t)t * QDIM + i + 32] : qp[(size_t)t * QDIM + i - 32];
    qo[(size_t)t * QDIM + i] = f2bf(x * c + xr * s);
  }
  for (int i = threadIdx.x; i < KVDIM; i += 256) {
    int d = i & 63, ii = d & 31;
    float ang = p * __expf(-(float)(2 * ii) * (LN_THETA / 64.f));
    float c = __cosf(ang), s = __sinf(ang);
    float x = kp[(size_t)t * KVDIM + i];
    float xr = (d < 32) ? -kp[(size_t)t * KVDIM + i + 32] : kp[(size_t)t * KVDIM + i - 32];
    ko[(size_t)t * KVDIM + i] = f2bf(x * c + xr * s);
    vo[(size_t)t * KVDIM + i] = f2bf(vp[(size_t)t * KVDIM + i]);
  }
}

// ---------------- flash attention (causal, GQA), WMMA bf16, 32-key tiles ----------------
// grid: B*NH*(S/64) blocks of 128 threads (4 waves, 16 queries/wave)
__global__ void __launch_bounds__(128) mx_attn(const unsigned short* __restrict__ Q,
                                               const unsigned short* __restrict__ K,
                                               const unsigned short* __restrict__ V,
                                               const int* __restrict__ amask,
                                               unsigned short* __restrict__ O) {
  __shared__ __align__(64) unsigned short Vt[HSZ * 32];   // V^T tile [hd][key], stride 32
  __shared__ __align__(64) unsigned short Pt[4][16 * 32]; // per-wave P tile [m][k]
  int blk = blockIdx.x;
  int qt = blk & 31;          // S/64 = 32
  int h = (blk >> 5) & 15;
  int b = blk >> 9;
  int kvh = h >> 2;
  int wvid = threadIdx.x >> 5, lane = threadIdx.x & 31;
  int ln = lane & 15, hf = lane >> 4;
  int qbase = qt * 64 + wvid * 16;

  AFrag qa[2];
  const unsigned short* qrow = Q + ((size_t)(b * SEQ + qbase + ln) * NHEAD + h) * HSZ;
#pragma unroll
  for (int st = 0; st < 2; ++st) {
    qa[st].h[0] = *(const v8bf*)(qrow + st * 32 + hf * 16);
    qa[st].h[1] = *(const v8bf*)(qrow + st * 32 + hf * 16 + 8);
  }

  float rm[8], rl[8];
#pragma unroll
  for (int r = 0; r < 8; ++r) { rm[r] = -3.0e38f; rl[r] = 0.f; }
  v8f acc[4];
#pragma unroll
  for (int i = 0; i < 4; ++i) acc[i] = zero8();

  int nkt = ((qt * 64 + 63) >> 5) + 1; // 32-key tiles, causal bound (uniform per block)
  for (int kt = 0; kt < nkt; ++kt) {
    int kb0 = kt * 32;
    __syncthreads();
    { // stage V^T: 32 keys x 64 hd; 128 threads x 16 elems
      int key = threadIdx.x & 31, hd0 = (threadIdx.x >> 5) * 16;
      const unsigned short* vrow =
          V + ((size_t)(b * SEQ + kb0 + key) * NKVH + kvh) * HSZ + hd0;
#pragma unroll
      for (int j = 0; j < 16; ++j) Vt[(hd0 + j) * 32 + key] = vrow[j];
    }
    __syncthreads();

    // scores for two 16-key halves, contraction over HD = 2 WMMA steps each
    v8f s0 = zero8(), s1 = zero8();
#pragma unroll
    for (int st = 0; st < 2; ++st) {
      BFrag kf;
      kf.v = *(const v16bf*)(K + ((size_t)(b * SEQ + kb0 + ln) * NKVH + kvh) * HSZ +
                             st * 32 + hf * 16);
      s0 = __builtin_amdgcn_wmma_f32_16x16x32_bf16(false, qa[st].v, false, kf.v,
                                                   (short)0, s0, false, false);
      kf.v = *(const v16bf*)(K + ((size_t)(b * SEQ + kb0 + 16 + ln) * NKVH + kvh) * HSZ +
                             st * 32 + hf * 16);
      s1 = __builtin_amdgcn_wmma_f32_16x16x32_bf16(false, qa[st].v, false, kf.v,
                                                   (short)0, s1, false, false);
    }

    int kg0 = kb0 + ln, kg1 = kb0 + 16 + ln;
    int mv0 = amask[b * SEQ + kg0], mv1 = amask[b * SEQ + kg1];
    float corr[8], p0[8], p1[8];
#pragma unroll
    for (int r = 0; r < 8; ++r) {
      int qg = qbase + hf * 8 + r;
      float sv0 = s0[r] * 0.125f; // 1/sqrt(64)
      float sv1 = s1[r] * 0.125f;
      if (!(kg0 <= qg && mv0)) sv0 = -3.0e38f;
      if (!(kg1 <= qg && mv1)) sv1 = -3.0e38f;
      float mx = fmaxf(sv0, sv1);
      for (int off = 1; off < 16; off <<= 1) mx = fmaxf(mx, __shfl_xor(mx, off, 32));
      float mnew = fmaxf(rm[r], mx);
      float c = __expf(rm[r] - mnew);
      float q0 = (sv0 <= -1e37f) ? 0.f : __expf(sv0 - mnew);
      float q1 = (sv1 <= -1e37f) ? 0.f : __expf(sv1 - mnew);
      float rs = q0 + q1;
      for (int off = 1; off < 16; off <<= 1) rs += __shfl_xor(rs, off, 32);
      rl[r] = rl[r] * c + rs;
      rm[r] = mnew;
      corr[r] = c;
      p0[r] = q0;
      p1[r] = q1;
    }
#pragma unroll
    for (int nt = 0; nt < 4; ++nt)
#pragma unroll
      for (int r = 0; r < 8; ++r) acc[nt][r] *= corr[r];

    // P (C-layout) -> LDS -> A-layout (full 16x32)
#pragma unroll
    for (int r = 0; r < 8; ++r) {
      Pt[wvid][(hf * 8 + r) * 32 + ln] = f2bf(p0[r]);
      Pt[wvid][(hf * 8 + r) * 32 + 16 + ln] = f2bf(p1[r]);
    }
    __syncthreads();
    AFrag pa;
    pa.h[0] = *(const v8bf*)&Pt[wvid][ln * 32 + hf * 16];
    pa.h[1] = *(const v8bf*)&Pt[wvid][ln * 32 + hf * 16 + 8];
#pragma unroll
    for (int nt = 0; nt < 4; ++nt) {
      BFrag vf;
      vf.v = *(const v16bf*)&Vt[(nt * 16 + ln) * 32 + hf * 16];
      acc[nt] = __builtin_amdgcn_wmma_f32_16x16x32_bf16(false, pa.v, false, vf.v,
                                                        (short)0, acc[nt], false, false);
    }
  }

#pragma unroll
  for (int nt = 0; nt < 4; ++nt)
#pragma unroll
    for (int r = 0; r < 8; ++r) {
      int qg = qbase + hf * 8 + r;
      float ov = acc[nt][r] / fmaxf(rl[r], 1e-30f);
      O[((size_t)(b * SEQ + qg) * NHEAD + h) * HSZ + nt * 16 + ln] = f2bf(ov);
    }
}

// ---------------- router (tiny GEMM, scalar) ----------------
__global__ void mx_router(const unsigned short* __restrict__ h2,
                          const unsigned short* __restrict__ gate_t,
                          float* __restrict__ logits) {
  int idx = blockIdx.x * 256 + threadIdx.x;
  if (idx >= TKN * NEXP) return;
  int t = idx >> 3, e = idx & 7;
  float s = 0.f;
  for (int i = 0; i < HDIM; ++i)
    s += bf2f(h2[(size_t)t * HDIM + i]) * bf2f(gate_t[(size_t)e * HDIM + i]);
  logits[idx] = s;
}

__global__ void mx_zero(int* p, int n) {
  int i = blockIdx.x * 256 + threadIdx.x;
  if (i < n) p[i] = 0;
}

__global__ void mx_topk(const float* __restrict__ logits, int* __restrict__ eidx,
                        float* __restrict__ ew, int* __restrict__ counts) {
  int t = blockIdx.x * 256 + threadIdx.x;
  if (t >= TKN) return;
  const float* L = logits + (size_t)t * NEXP;
  int b0 = 0; float v0 = L[0];
  for (int e = 1; e < NEXP; ++e) if (L[e] > v0) { v0 = L[e]; b0 = e; }
  int b1 = (b0 == 0) ? 1 : 0; float v1 = L[b1];
  for (int e = 0; e < NEXP; ++e)
    if (e != b0 && L[e] > v1) { v1 = L[e]; b1 = e; }
  float e1 = __expf(v1 - v0);
  ew[t * 2] = 1.f / (1.f + e1);
  ew[t * 2 + 1] = e1 / (1.f + e1);
  eidx[t * 2] = b0; eidx[t * 2 + 1] = b1;
  atomicAdd(&counts[b0], 1); atomicAdd(&counts[b1], 1);
}

__global__ void mx_scan(const int* __restrict__ counts, int* __restrict__ offs,
                        int* __restrict__ cursor) {
  if (threadIdx.x == 0 && blockIdx.x == 0) {
    int s = 0;
    for (int e = 0; e < NEXP; ++e) { offs[e] = s; cursor[e] = 0; s += counts[e]; }
    offs[NEXP] = s;
  }
}

__global__ void mx_assign(const int* __restrict__ eidx, const int* __restrict__ offs,
                          int* __restrict__ cursor, int* __restrict__ toks,
                          int* __restrict__ slot_of) {
  int t = blockIdx.x * 256 + threadIdx.x;
  if (t >= TKN) return;
  for (int k = 0; k < 2; ++k) {
    int e = eidx[t * 2 + k];
    int pos = atomicAdd(&cursor[e], 1);
    int slot = offs[e] + pos;
    toks[slot] = t;
    slot_of[t * 2 + k] = slot;
  }
}

// ---------------- MoE gathered GEMM 1: inter = silu(h@w1)*(h@w3) ----------------
__global__ void __launch_bounds__(256) mx_moe_gemm13(
    const unsigned short* __restrict__ Hs, const unsigned short* __restrict__ W1t,
    const unsigned short* __restrict__ W3t, const int* __restrict__ offs,
    const int* __restrict__ toks, unsigned short* __restrict__ inter) {
  __shared__ __align__(64) unsigned short As[2][128 * 64];
  __shared__ __align__(64) unsigned short B1[2][64 * 64];
  __shared__ __align__(64) unsigned short B3[2][64 * 64];
  int e = blockIdx.z;
  int beg = offs[e], cnt = offs[e + 1] - beg;
  int m0 = blockIdx.x * 128;
  if (m0 >= cnt) return;
  int n0 = blockIdx.y * 64;
  const unsigned short* W1e = W1t + (size_t)e * IDIM * HDIM;
  const unsigned short* W3e = W3t + (size_t)e * IDIM * HDIM;
  int tid = threadIdx.x, wvid = tid >> 5, lane = tid & 31;
  int ln = lane & 15, hf = lane >> 4;
  v8f a1[4], a3[4];
#pragma unroll
  for (int i = 0; i < 4; ++i) { a1[i] = zero8(); a3[i] = zero8(); }

  const int ar = tid >> 1, ac = (tid & 1) * 32;
  const int br = tid >> 2, bc = (tid & 3) * 16;
  int arowg = m0 + ar; if (arowg >= cnt) arowg = cnt - 1;
  const unsigned short* gA = Hs + (size_t)toks[beg + arowg] * HDIM + ac;
  const unsigned short* g1 = W1e + (size_t)(n0 + br) * HDIM + bc;
  const unsigned short* g3 = W3e + (size_t)(n0 + br) * HDIM + bc;
  auto stage = [&](int buf, int k0) {
#pragma unroll
    for (int j = 0; j < 4; ++j)
      mx_cp16(gA + k0 + j * 8, &As[buf][ar * 64 + ac + j * 8]);
    mx_cp16(g1 + k0, &B1[buf][br * 64 + bc]);
    mx_cp16(g1 + k0 + 8, &B1[buf][br * 64 + bc + 8]);
    mx_cp16(g3 + k0, &B3[buf][br * 64 + bc]);
    mx_cp16(g3 + k0 + 8, &B3[buf][br * 64 + bc + 8]);
  };
  auto compute = [&](int buf) {
    int arow = wvid * 16 + ln;
#pragma unroll
    for (int kk = 0; kk < 2; ++kk) {
      AFrag a;
      a.h[0] = *(const v8bf*)&As[buf][arow * 64 + kk * 32 + hf * 16];
      a.h[1] = *(const v8bf*)&As[buf][arow * 64 + kk * 32 + hf * 16 + 8];
#pragma unroll
      for (int nt = 0; nt < 4; ++nt) {
        BFrag b;
        b.v = *(const v16bf*)&B1[buf][(nt * 16 + ln) * 64 + kk * 32 + hf * 16];
        a1[nt] = __builtin_amdgcn_wmma_f32_16x16x32_bf16(false, a.v, false, b.v,
                                                         (short)0, a1[nt], false, false);
        b.v = *(const v16bf*)&B3[buf][(nt * 16 + ln) * 64 + kk * 32 + hf * 16];
        a3[nt] = __builtin_amdgcn_wmma_f32_16x16x32_bf16(false, a.v, false, b.v,
                                                         (short)0, a3[nt], false, false);
      }
    }
  };
  stage(0, 0);
  const int nk = HDIM / 64;
  for (int ki = 0; ki < nk - 1; ++ki) {
    stage((ki + 1) & 1, (ki + 1) * 64);
    mx_wait_stage<8>();
    __syncthreads();
    compute(ki & 1);
    __syncthreads();
  }
  mx_wait_stage<0>();
  __syncthreads();
  compute((nk - 1) & 1);

#pragma unroll
  for (int nt = 0; nt < 4; ++nt)
#pragma unroll
    for (int r = 0; r < 8; ++r) {
      int m = wvid * 16 + hf * 8 + r;
      if (m0 + m < cnt) {
        float g = a1[nt][r];
        float val = (g / (1.f + __expf(-g))) * a3[nt][r];
        inter[(size_t)(beg + m0 + m) * IDIM + n0 + nt * 16 + ln] = f2bf(val);
      }
    }
}

// ---------------- MoE GEMM 2: aout = inter @ w2[e] ----------------
__global__ void __launch_bounds__(256) mx_moe_gemm2(
    const unsigned short* __restrict__ inter, const unsigned short* __restrict__ W2t,
    const int* __restrict__ offs, float* __restrict__ aout) {
  __shared__ __align__(64) unsigned short As[2][128 * 64];
  __shared__ __align__(64) unsigned short Bs[2][64 * 64];
  int e = blockIdx.z;
  int beg = offs[e], cnt = offs[e + 1] - beg;
  int m0 = blockIdx.x * 128;
  if (m0 >= cnt) return;
  int n0 = blockIdx.y * 64;
  const unsigned short* W2e = W2t + (size_t)e * HDIM * IDIM;
  int tid = threadIdx.x, wvid = tid >> 5, lane = tid & 31;
  int ln = lane & 15, hf = lane >> 4;
  v8f acc[4];
#pragma unroll
  for (int i = 0; i < 4; ++i) acc[i] = zero8();

  const int ar = tid >> 1, ac = (tid & 1) * 32;
  const int br = tid >> 2, bc = (tid & 3) * 16;
  int arowg = m0 + ar; if (arowg >= cnt) arowg = cnt - 1;
  const unsigned short* gA = inter + (size_t)(beg + arowg) * IDIM + ac;
  const unsigned short* gB = W2e + (size_t)(n0 + br) * IDIM + bc;
  auto stage = [&](int buf, int k0) {
#pragma unroll
    for (int j = 0; j < 4; ++j)
      mx_cp16(gA + k0 + j * 8, &As[buf][ar * 64 + ac + j * 8]);
    mx_cp16(gB + k0, &Bs[buf][br * 64 + bc]);
    mx_cp16(gB + k0 + 8, &Bs[buf][br * 64 + bc + 8]);
  };
  auto compute = [&](int buf) {
    int arow = wvid * 16 + ln;
#pragma unroll
    for (int kk = 0; kk < 2; ++kk) {
      AFrag a;
      a.h[0] = *(const v8bf*)&As[buf][arow * 64 + kk * 32 + hf * 16];
      a.h[1] = *(const v8bf*)&As[buf][arow * 64 + kk * 32 + hf * 16 + 8];
#pragma unroll
      for (int nt = 0; nt < 4; ++nt) {
        BFrag b;
        b.v = *(const v16bf*)&Bs[buf][(nt * 16 + ln) * 64 + kk * 32 + hf * 16];
        acc[nt] = __builtin_amdgcn_wmma_f32_16x16x32_bf16(false, a.v, false, b.v,
                                                          (short)0, acc[nt], false, false);
      }
    }
  };
  stage(0, 0);
  const int nk = IDIM / 64;
  for (int ki = 0; ki < nk - 1; ++ki) {
    stage((ki + 1) & 1, (ki + 1) * 64);
    mx_wait_stage<6>();
    __syncthreads();
    compute(ki & 1);
    __syncthreads();
  }
  mx_wait_stage<0>();
  __syncthreads();
  compute((nk - 1) & 1);

#pragma unroll
  for (int nt = 0; nt < 4; ++nt)
#pragma unroll
    for (int r = 0; r < 8; ++r) {
      int m = wvid * 16 + hf * 8 + r;
      if (m0 + m < cnt)
        aout[(size_t)(beg + m0 + m) * HDIM + n0 + nt * 16 + ln] = acc[nt][r];
    }
}

// ---------------- final combine: x = x1 + w0*eo0 + w1*eo1 ----------------
__global__ void mx_combine(const float* __restrict__ x1, const float* __restrict__ aout,
                           const int* __restrict__ slot_of, const float* __restrict__ ew,
                           float* __restrict__ out) {
  int idx = blockIdx.x * 256 + threadIdx.x;
  if (idx >= TKN * HDIM) return;
  int t = idx >> 10, hcol = idx & 1023;
  int s0 = slot_of[t * 2], s1 = slot_of[t * 2 + 1];
  out[idx] = x1[idx] + ew[t * 2] * aout[(size_t)s0 * HDIM + hcol] +
             ew[t * 2 + 1] * aout[(size_t)s1 * HDIM + hcol];
}

// ---------------- host orchestration ----------------
extern "C" void kernel_launch(void* const* d_in, const int* in_sizes, int n_in,
                              void* d_out, int out_size, void* d_ws, size_t ws_size,
                              hipStream_t stream) {
  (void)in_sizes; (void)n_in; (void)out_size; (void)ws_size;
  const float* hidden = (const float*)d_in[0];
  const int* amask    = (const int*)d_in[1];
  const int* pos      = (const int*)d_in[2];
  const float* n1w    = (const float*)d_in[3];
  const float* n2w    = (const float*)d_in[4];
  const float* wq     = (const float*)d_in[5];
  const float* wk     = (const float*)d_in[6];
  const float* wv     = (const float*)d_in[7];
  const float* wo     = (const float*)d_in[8];
  const float* gw     = (const float*)d_in[9];
  const float* w1     = (const float*)d_in[10];
  const float* w3     = (const float*)d_in[11];
  const float* w2     = (const float*)d_in[12];
  float* out_x = (float*)d_out;
  float* out_logits = out_x + (size_t)TKN * HDIM;

  char* p = (char*)d_ws;
  auto alloc = [&](size_t bytes) -> char* {
    char* r = p; p += (bytes + 255) & ~(size_t)255; return r;
  };
  unsigned short* wq_t  = (unsigned short*)alloc((size_t)QDIM * HDIM * 2);
  unsigned short* wk_t  = (unsigned short*)alloc((size_t)KVDIM * HDIM * 2);
  unsigned short* wv_t  = (unsigned short*)alloc((size_t)KVDIM * HDIM * 2);
  unsigned short* wo_t  = (unsigned short*)alloc((size_t)HDIM * QDIM * 2);
  unsigned short* gw_t  = (unsigned short*)alloc((size_t)NEXP * HDIM * 2);
  unsigned short* w1_t  = (unsigned short*)alloc((size_t)NEXP * IDIM * HDIM * 2);
  unsigned short* w3_t  = (unsigned short*)alloc((size_t)NEXP * IDIM * HDIM * 2);
  unsigned short* w2_t  = (unsigned short*)alloc((size_t)NEXP * HDIM * IDIM * 2);
  unsigned short* hnorm = (unsigned short*)alloc((size_t)TKN * HDIM * 2);
  unsigned short* qb    = (unsigned short*)alloc((size_t)TKN * QDIM * 2);
  unsigned short* kb    = (unsigned short*)alloc((size_t)TKN * KVDIM * 2);
  unsigned short* vb    = (unsigned short*)alloc((size_t)TKN * KVDIM * 2);
  unsigned short* atto  = (unsigned short*)alloc((size_t)TKN * QDIM * 2);
  unsigned short* h2    = (unsigned short*)alloc((size_t)TKN * HDIM * 2);
  unsigned short* inter = (unsigned short*)alloc((size_t)NASSIGN * IDIM * 2);
  float* qproj = (float*)alloc((size_t)TKN * QDIM * 4);
  float* kproj = (float*)alloc((size_t)TKN * KVDIM * 4);
  float* vproj = (float*)alloc((size_t)TKN * KVDIM * 4);
  float* x1    = (float*)alloc((size_t)TKN * HDIM * 4);
  float* aout  = (float*)alloc((size_t)NASSIGN * HDIM * 4);
  float* ew    = (float*)alloc((size_t)NASSIGN * 4);
  int* eidx    = (int*)alloc((size_t)NASSIGN * 4);
  int* slot_of = (int*)alloc((size_t)NASSIGN * 4);
  int* toks    = (int*)alloc((size_t)NASSIGN * 4);
  int* counts  = (int*)alloc(64 * 4); // counts[8], offs[9], cursor[8]
  int* offs    = counts + 8;
  int* cursor  = counts + 24;

  // weight conversion + transpose (fp32 -> bf16 [N][K])
  mx_convT<<<dim3((HDIM * QDIM + 255) / 256, 1, 1), 256, 0, stream>>>(wq, wq_t, HDIM, QDIM);
  mx_convT<<<dim3((HDIM * KVDIM + 255) / 256, 1, 1), 256, 0, stream>>>(wk, wk_t, HDIM, KVDIM);
  mx_convT<<<dim3((HDIM * KVDIM + 255) / 256, 1, 1), 256, 0, stream>>>(wv, wv_t, HDIM, KVDIM);
  mx_convT<<<dim3((QDIM * HDIM + 255) / 256, 1, 1), 256, 0, stream>>>(wo, wo_t, QDIM, HDIM);
  mx_convT<<<dim3((HDIM * NEXP + 255) / 256, 1, 1), 256, 0, stream>>>(gw, gw_t, HDIM, NEXP);
  mx_convT<<<dim3((HDIM * IDIM + 255) / 256, 1, NEXP), 256, 0, stream>>>(w1, w1_t, HDIM, IDIM);
  mx_convT<<<dim3((HDIM * IDIM + 255) / 256, 1, NEXP), 256, 0, stream>>>(w3, w3_t, HDIM, IDIM);
  mx_convT<<<dim3((IDIM * HDIM + 255) / 256, 1, NEXP), 256, 0, stream>>>(w2, w2_t, IDIM, HDIM);

  // attention block
  mx_rmsnorm<<<TKN, 256, 0, stream>>>(hidden, n1w, hnorm);
  mx_gemm<<<dim3(TKN / 128, QDIM / 64), 256, 0, stream>>>(hnorm, wq_t, qproj, nullptr, TKN, QDIM, HDIM);
  mx_gemm<<<dim3(TKN / 128, KVDIM / 64), 256, 0, stream>>>(hnorm, wk_t, kproj, nullptr, TKN, KVDIM, HDIM);
  mx_gemm<<<dim3(TKN / 128, KVDIM / 64), 256, 0, stream>>>(hnorm, wv_t, vproj, nullptr, TKN, KVDIM, HDIM);
  mx_rope<<<TKN, 256, 0, stream>>>(qproj, kproj, vproj, pos, qb, kb, vb);
  mx_attn<<<BATCH * NHEAD * (SEQ / 64), 128, 0, stream>>>(qb, kb, vb, amask, atto);
  mx_gemm<<<dim3(TKN / 128, HDIM / 64), 256, 0, stream>>>(atto, wo_t, x1, hidden, TKN, HDIM, QDIM);

  // MoE block
  mx_rmsnorm<<<TKN, 256, 0, stream>>>(x1, n2w, h2);
  mx_router<<<(TKN * NEXP + 255) / 256, 256, 0, stream>>>(h2, gw_t, out_logits);
  mx_zero<<<1, 64, 0, stream>>>(counts, 41);
  mx_topk<<<(TKN + 255) / 256, 256, 0, stream>>>(out_logits, eidx, ew, counts);
  mx_scan<<<1, 1, 0, stream>>>(counts, offs, cursor);
  mx_assign<<<(TKN + 255) / 256, 256, 0, stream>>>(eidx, offs, cursor, toks, slot_of);
  mx_moe_gemm13<<<dim3(32, IDIM / 64, NEXP), 256, 0, stream>>>(h2, w1_t, w3_t, offs, toks, inter);
  mx_moe_gemm2<<<dim3(32, HDIM / 64, NEXP), 256, 0, stream>>>(inter, w2_t, offs, aout);
  mx_combine<<<(TKN * HDIM + 255) / 256, 256, 0, stream>>>(x1, aout, slot_of, ew, out_x);
}
